// MultiHeadAttention_40114994544681
// MI455X (gfx1250) — matrix operand order repaired and checked by reading, not compiled, not run on hardware
//
#include <hip/hip_runtime.h>

// ---------------------------------------------------------------------------
// MHA forward for MI455X (gfx1250): bf16 WMMA GEMMs + async-LDS flash attention.
// ---------------------------------------------------------------------------

typedef __attribute__((ext_vector_type(16))) __bf16 v16bf;
typedef __attribute__((ext_vector_type(8)))  __bf16 v8bf;
typedef __attribute__((ext_vector_type(8)))  float  v8f;
typedef __attribute__((ext_vector_type(4)))  int    v4i;

#define BSZ 4
#define TSZ 2048
#define CSZ 1024
#define HN  16
#define HD  64
#define NEG_INF (-__builtin_inff())

#if defined(__AMDGCN__) && \
    __has_builtin(__builtin_amdgcn_global_load_async_to_lds_b128) && \
    __has_builtin(__builtin_amdgcn_s_wait_asynccnt)
#define FA_ASYNC 1
#define AS1 __attribute__((address_space(1)))
#define AS3 __attribute__((address_space(3)))
#else
#define FA_ASYNC 0
#endif

static __device__ __forceinline__ v8f wmma_bf16(v16bf a, v16bf b, v8f c) {
  // D = A(16x32 bf16) * B(32x16 bf16) + C(16x16 f32)
  return __builtin_amdgcn_wmma_f32_16x16x32_bf16(
      /*neg_a=*/false, a, /*neg_b=*/false, b,
      /*c_mod=*/(short)0, c, /*reuse_a=*/false, /*reuse_b=*/false);
}

// A-fragment (16x32, 16-bit): lane holds row (lane&15); K elems j -> K=16*ks+j.
// base = &row[16*ks]; second chunk at +8 elements.
static __device__ __forceinline__ v16bf make_afrag(const __bf16* base) {
  v8bf lo = *reinterpret_cast<const v8bf*>(base);
  v8bf hi = *reinterpret_cast<const v8bf*>(base + 8);
  v16bf a;
#pragma unroll
  for (int i = 0; i < 8; ++i) { a[i] = lo[i]; a[i + 8] = hi[i]; }
  return a;
}

// ---------------------------------------------------------------------------
// fp32 -> bf16 conversion (4 elems/thread; all sizes are multiples of 4)
// ---------------------------------------------------------------------------
__global__ void __launch_bounds__(256)
cvt_f32_bf16(const float* __restrict__ in, __bf16* __restrict__ out, size_t n) {
  size_t i = ((size_t)blockIdx.x * blockDim.x + threadIdx.x) * 4;
  if (i + 3 < n) {
    float4 v = *reinterpret_cast<const float4*>(in + i);
    out[i + 0] = (__bf16)v.x;
    out[i + 1] = (__bf16)v.y;
    out[i + 2] = (__bf16)v.z;
    out[i + 3] = (__bf16)v.w;
  }
}

// ---------------------------------------------------------------------------
// QKV projection: Y[r,d] = sum_c X[r,c] * Wqkv[d,c]   (r=B*T rows, d=3C cols)
// Scatter: d -> head h=d/192, which=(d%192)/64 (0=q,1=k,2=v), c=d%64.
//   Q,K : [B,H,T,hd] bf16   (Q pre-scaled by hd^-0.5 = 0.125)
//   Vt  : [B,H,hd,T] bf16   (transposed so PV B-frags are contiguous)
// One wave computes a 64x64 output tile (4x4 WMMA tiles).
// ---------------------------------------------------------------------------
__global__ void __launch_bounds__(128)
qkv_gemm(const __bf16* __restrict__ X, const __bf16* __restrict__ W,
         __bf16* __restrict__ Q, __bf16* __restrict__ K,
         __bf16* __restrict__ Vt) {
  const int lane = threadIdx.x & 31;
  const int n    = lane & 15;
  const int ks   = lane >> 4;
  const int wg   = blockIdx.x * 4 + (threadIdx.x >> 5);
  const int m0   = (wg & 127) * 64;   // 128 M-tiles
  const int n0   = (wg >> 7) * 64;    // 48 N-tiles

  v8f acc[4][4];
#pragma unroll
  for (int i = 0; i < 4; ++i)
#pragma unroll
    for (int j = 0; j < 4; ++j)
#pragma unroll
      for (int r = 0; r < 8; ++r) acc[i][j][r] = 0.0f;

  for (int c0 = 0; c0 < CSZ; c0 += 32) {
    v16bf af[4], bfr[4];
#pragma unroll
    for (int im = 0; im < 4; ++im)
      af[im] = make_afrag(X + (size_t)(m0 + 16 * im + n) * CSZ + c0 + 16 * ks);
#pragma unroll
    for (int in = 0; in < 4; ++in)
      bfr[in] = *reinterpret_cast<const v16bf*>(
          W + (size_t)(n0 + 16 * in + n) * CSZ + c0 + 16 * ks);
#pragma unroll
    for (int im = 0; im < 4; ++im)
#pragma unroll
      for (int in = 0; in < 4; ++in)
        acc[im][in] = wmma_bf16(af[im], bfr[in], acc[im][in]);
  }

#pragma unroll
  for (int in = 0; in < 4; ++in) {
    const int dd    = n0 + 16 * in + n;
    const int h     = dd / 192;
    const int rem   = dd % 192;
    const int which = rem >> 6;
    const int cc    = rem & 63;
#pragma unroll
    for (int im = 0; im < 4; ++im) {
#pragma unroll
      for (int r = 0; r < 8; ++r) {
        const int rr = m0 + 16 * im + r + 8 * ks;  // global row = b*T + t
        const int b  = rr >> 11;
        const int t  = rr & 2047;
        const size_t bh = (size_t)(b * HN + h);
        const float v = acc[im][in][r];
        if (which == 0)
          Q[(bh * TSZ + t) * HD + cc] = (__bf16)(v * 0.125f);
        else if (which == 1)
          K[(bh * TSZ + t) * HD + cc] = (__bf16)v;
        else
          Vt[(bh * HD + cc) * TSZ + t] = (__bf16)v;
      }
    }
  }
}

// ---------------------------------------------------------------------------
// Causal flash attention. 4-wave block owns 64 query rows of one (b,h); each
// wave owns 16 of them.  K/V k-tiles (32 kv rows) are staged into LDS with
// GLOBAL_LOAD_ASYNC_TO_LDS_B128 (double-buffered, ASYNCcnt-tracked) and shared
// by all 4 waves.  Online softmax keeps row stats per-lane matching the C/D
// VGPR layout (row = r + 8*(lane>=16), col = lane&15); P is re-laid out into
// A-fragment form through a per-wave LDS tile (DS ops are in-order per wave).
// Output Ao: [B,T,H,hd] bf16  (== [B*T, C] row-major for the out-projection).
// ---------------------------------------------------------------------------
__global__ void __launch_bounds__(128)
flash_attn(const __bf16* __restrict__ Q, const __bf16* __restrict__ K,
           const __bf16* __restrict__ Vt, __bf16* __restrict__ Ao) {
  const int tid  = threadIdx.x;
  const int lane = tid & 31;
  const int w    = tid >> 5;          // wave in block (0..3)
  const int n  = lane & 15;
  const int ks = lane >> 4;
  const int qb = blockIdx.x & 31;     // 32 q-blocks of 64 rows
  const int bh = blockIdx.x >> 5;     // 64 (b,h) pairs
  const int b  = bh >> 4;
  const int h  = bh & 15;
  const int q0blk = qb * 64;
  const int q0    = q0blk + w * 16;   // this wave's 16 query rows

  const __bf16* Qh = Q  + (size_t)bh * TSZ * HD;
  const __bf16* Kh = K  + (size_t)bh * TSZ * HD;
  const __bf16* Vh = Vt + (size_t)bh * HD  * TSZ;

  __shared__ alignas(128) __bf16 Ktile[2][32 * HD];  // [buf][kvrow*64 + c]
  __shared__ alignas(128) __bf16 Vtile[2][HD * 32];  // [buf][c*32 + tloc]
  __shared__ alignas(64)  __bf16 Pl[4][16 * 32];     // per-wave P tile

  // Stage one 32-row K/V tile: 256 b128 chunks each, 2+2 chunks per thread.
  auto stage_issue = [&](int k0s, int buf) {
#if FA_ASYNC
#pragma unroll
    for (int j = 0; j < 2; ++j) {
      const int i = tid * 2 + j;  // 0..255
      __builtin_amdgcn_global_load_async_to_lds_b128(
          (AS1 v4i*)(Kh + (size_t)(k0s + (i >> 3)) * HD + (i & 7) * 8),
          (AS3 v4i*)&Ktile[buf][i * 8], 0, 0);
    }
#pragma unroll
    for (int j = 0; j < 2; ++j) {
      const int i = tid * 2 + j;
      __builtin_amdgcn_global_load_async_to_lds_b128(
          (AS1 v4i*)(Vh + (size_t)(i >> 2) * TSZ + k0s + (i & 3) * 8),
          (AS3 v4i*)&Vtile[buf][i * 8], 0, 0);
    }
#else
#pragma unroll
    for (int j = 0; j < 2; ++j) {
      const int i = tid * 2 + j;
      *reinterpret_cast<v8bf*>(&Ktile[buf][i * 8]) =
          *reinterpret_cast<const v8bf*>(
              Kh + (size_t)(k0s + (i >> 3)) * HD + (i & 7) * 8);
    }
#pragma unroll
    for (int j = 0; j < 2; ++j) {
      const int i = tid * 2 + j;
      *reinterpret_cast<v8bf*>(&Vtile[buf][i * 8]) =
          *reinterpret_cast<const v8bf*>(
              Vh + (size_t)(i >> 2) * TSZ + k0s + (i & 3) * 8);
    }
#endif
  };

  // Q fragments (16 rows x 64 c, two 32-wide K-steps) live in registers.
  v16bf qf[2];
#pragma unroll
  for (int kc = 0; kc < 2; ++kc)
    qf[kc] = make_afrag(Qh + (size_t)(q0 + n) * HD + 32 * kc + 16 * ks);

  float mrow[8], lrow[8];
  v8f o[4];
#pragma unroll
  for (int r = 0; r < 8; ++r) { mrow[r] = NEG_INF; lrow[r] = 0.0f; }
#pragma unroll
  for (int ct = 0; ct < 4; ++ct)
#pragma unroll
    for (int r = 0; r < 8; ++r) o[ct][r] = 0.0f;

  const int kend = q0blk + 64;        // block processes k0 in [0, kend)
  stage_issue(0, 0);

  for (int k0 = 0; k0 < kend; k0 += 32) {
    const int buf = (k0 >> 5) & 1;
    const bool has_next = (k0 + 32) < kend;
    if (has_next) stage_issue(k0 + 32, buf ^ 1);
#if FA_ASYNC
    if (has_next) __builtin_amdgcn_s_wait_asynccnt(4);  // current stage landed
    else          __builtin_amdgcn_s_wait_asynccnt(0);
#endif
    __syncthreads();  // all waves' K/V tile data visible in LDS

    // Causal guard: this wave's rows only need k <= q0+15 (wave-uniform, so
    // EXEC stays all-ones inside -> WMMA legal).
    if (k0 < q0 + 16) {
      // S = Q K^T  (scale already folded into Q)
      v8f s[2];
#pragma unroll
      for (int nt = 0; nt < 2; ++nt)
#pragma unroll
        for (int r = 0; r < 8; ++r) s[nt][r] = 0.0f;
#pragma unroll
      for (int nt = 0; nt < 2; ++nt) {
#pragma unroll
        for (int kc = 0; kc < 2; ++kc) {
          v16bf kf = *reinterpret_cast<const v16bf*>(
              &Ktile[buf][(16 * nt + n) * HD + 32 * kc + 16 * ks]);
          s[nt] = wmma_bf16(qf[kc], kf, s[nt]);
        }
      }

      // causal mask + row max (reduce across the 16 lanes holding each row)
      float rmax[8];
#pragma unroll
      for (int r = 0; r < 8; ++r) {
        const int qg = q0 + r + 8 * ks;
#pragma unroll
        for (int nt = 0; nt < 2; ++nt) {
          const int kg = k0 + 16 * nt + n;
          if (kg > qg) s[nt][r] = NEG_INF;
        }
        float mx = fmaxf(s[0][r], s[1][r]);
#pragma unroll
        for (int off = 1; off < 16; off <<= 1)
          mx = fmaxf(mx, __shfl_xor(mx, off, 32));
        rmax[r] = mx;
      }

      // online softmax update
#pragma unroll
      for (int r = 0; r < 8; ++r) {
        const float mn    = fmaxf(mrow[r], rmax[r]);
        const float alpha = __expf(mrow[r] - mn);
        mrow[r] = mn;
        const float p0 = __expf(s[0][r] - mn);
        const float p1 = __expf(s[1][r] - mn);
        s[0][r] = p0; s[1][r] = p1;
        float rs = p0 + p1;
#pragma unroll
        for (int off = 1; off < 16; off <<= 1)
          rs += __shfl_xor(rs, off, 32);
        lrow[r] = lrow[r] * alpha + rs;
#pragma unroll
        for (int ct = 0; ct < 4; ++ct) o[ct][r] *= alpha;
      }

      // P (f32 C-layout) -> per-wave LDS tile -> A-fragment.  DS ops are
      // in-order within a wave; the asm blocks only stop compiler reordering.
#pragma unroll
      for (int r = 0; r < 8; ++r) {
        Pl[w][(r + 8 * ks) * 32 + n]      = (__bf16)s[0][r];
        Pl[w][(r + 8 * ks) * 32 + 16 + n] = (__bf16)s[1][r];
      }
      asm volatile("" ::: "memory");
      v16bf pf = make_afrag(&Pl[w][n * 32 + 16 * ks]);

      // O += P V   (Vtile rows are hd-channels, contiguous in t)
#pragma unroll
      for (int ct = 0; ct < 4; ++ct) {
        v16bf vf = *reinterpret_cast<const v16bf*>(
            &Vtile[buf][(16 * ct + n) * 32 + 16 * ks]);
        o[ct] = wmma_bf16(pf, vf, o[ct]);
      }
    }
    __syncthreads();  // everyone done reading buf before it is re-staged
  }

  // normalize and write [B,T,H,hd]
#pragma unroll
  for (int r = 0; r < 8; ++r) {
    const float inv = 1.0f / lrow[r];
    const int t = q0 + r + 8 * ks;
#pragma unroll
    for (int ct = 0; ct < 4; ++ct) {
      const int cc = 16 * ct + n;
      Ao[(((size_t)b * TSZ + t) * HN + h) * HD + cc] = (__bf16)(o[ct][r] * inv);
    }
  }
}

// ---------------------------------------------------------------------------
// Output projection: out[r,d] = sum_c Ao[r,c] * Wproj[d,c] + bias[d]  (fp32 out)
// ---------------------------------------------------------------------------
__global__ void __launch_bounds__(128)
proj_gemm(const __bf16* __restrict__ A, const __bf16* __restrict__ W,
          const float* __restrict__ bias, float* __restrict__ out) {
  const int lane = threadIdx.x & 31;
  const int n  = lane & 15;
  const int ks = lane >> 4;
  const int wg = blockIdx.x * 4 + (threadIdx.x >> 5);
  const int m0 = (wg & 127) * 64;   // 128 M-tiles
  const int n0 = (wg >> 7) * 64;    // 16 N-tiles

  v8f acc[4][4];
#pragma unroll
  for (int i = 0; i < 4; ++i)
#pragma unroll
    for (int j = 0; j < 4; ++j)
#pragma unroll
      for (int r = 0; r < 8; ++r) acc[i][j][r] = 0.0f;

  for (int c0 = 0; c0 < CSZ; c0 += 32) {
    v16bf af[4], bfr[4];
#pragma unroll
    for (int im = 0; im < 4; ++im)
      af[im] = make_afrag(A + (size_t)(m0 + 16 * im + n) * CSZ + c0 + 16 * ks);
#pragma unroll
    for (int in = 0; in < 4; ++in)
      bfr[in] = *reinterpret_cast<const v16bf*>(
          W + (size_t)(n0 + 16 * in + n) * CSZ + c0 + 16 * ks);
#pragma unroll
    for (int im = 0; im < 4; ++im)
#pragma unroll
      for (int in = 0; in < 4; ++in)
        acc[im][in] = wmma_bf16(af[im], bfr[in], acc[im][in]);
  }

#pragma unroll
  for (int in = 0; in < 4; ++in) {
    const int dd = n0 + 16 * in + n;
    const float bv = bias[dd];
#pragma unroll
    for (int im = 0; im < 4; ++im)
#pragma unroll
      for (int r = 0; r < 8; ++r) {
        const int rr = m0 + 16 * im + r + 8 * ks;
        out[(size_t)rr * CSZ + dd] = acc[im][in][r] + bv;
      }
  }
}

// ---------------------------------------------------------------------------
extern "C" void kernel_launch(void* const* d_in, const int* in_sizes, int n_in,
                              void* d_out, int out_size, void* d_ws, size_t ws_size,
                              hipStream_t stream) {
  const float* x      = (const float*)d_in[0];
  const float* w_qkv  = (const float*)d_in[1];
  const float* w_proj = (const float*)d_in[2];
  const float* b_proj = (const float*)d_in[3];
  float* out = (float*)d_out;

  const size_t nx  = (size_t)BSZ * TSZ * CSZ;  // 8M
  const size_t nwq = (size_t)3 * CSZ * CSZ;    // 3M
  const size_t nwp = (size_t)CSZ * CSZ;        // 1M

  __bf16* xb  = (__bf16*)d_ws;   // bf16 workspace: 44M elems = 88 MB total
  __bf16* wqb = xb  + nx;
  __bf16* wpb = wqb + nwq;
  __bf16* Qs  = wpb + nwp;
  __bf16* Ks  = Qs  + nx;
  __bf16* Vts = Ks  + nx;
  __bf16* Aos = Vts + nx;

  cvt_f32_bf16<<<(unsigned)(nx / 4 / 256), 256, 0, stream>>>(x, xb, nx);
  cvt_f32_bf16<<<(unsigned)(nwq / 4 / 256), 256, 0, stream>>>(w_qkv, wqb, nwq);
  cvt_f32_bf16<<<(unsigned)(nwp / 4 / 256), 256, 0, stream>>>(w_proj, wpb, nwp);

  // 8192x3072 output, 64x64 per wave, 4 waves/block -> 1536 blocks
  qkv_gemm<<<1536, 128, 0, stream>>>(xb, wqb, Qs, Ks, Vts);

  // 64 (b,h) * 32 q-blocks of 64 rows = 2048 four-wave blocks
  flash_attn<<<2048, 128, 0, stream>>>(Qs, Ks, Vts, Aos);

  // 8192x1024 output -> 2048 waves -> 512 blocks
  proj_gemm<<<512, 128, 0, stream>>>(Aos, wpb, b_proj, out);
}
